// LogicMetaLerpLayer_60215441490000
// MI455X (gfx1250) — matrix operand order repaired and checked by reading, not compiled, not run on hardware
//
#include <hip/hip_runtime.h>
#include <hip/hip_bf16.h>
#include <math.h>

#define WIDTH 128
#define N_NODE 1024
#define N_REL 12

typedef __attribute__((ext_vector_type(16))) __bf16 v16bf;
typedef __attribute__((ext_vector_type(8)))  __bf16 v8bf;
typedef __attribute__((ext_vector_type(8)))  float  v8f;

// ---------------------------------------------------------------------------
// softmax over axis 0 of a 128x128 matrix (per column), one block per column
// ---------------------------------------------------------------------------
__global__ void k_softmax_col128(const float* __restrict__ W, float* __restrict__ out) {
    __shared__ float red[128];
    const int w = blockIdx.x;
    const int j = threadIdx.x;
    const float v = W[j * 128 + w];
    red[j] = v;
    __syncthreads();
    for (int s = 64; s > 0; s >>= 1) {
        if (j < s) red[j] = fmaxf(red[j], red[j + s]);
        __syncthreads();
    }
    const float m = red[0];
    __syncthreads();
    const float e = __expf(v - m);
    red[j] = e;
    __syncthreads();
    for (int s = 64; s > 0; s >>= 1) {
        if (j < s) red[j] += red[j + s];
        __syncthreads();
    }
    out[j * 128 + w] = e / red[0];
}

// ---------------------------------------------------------------------------
// softmax over axis 1 (rows) of a 128 x ncol matrix, one thread per row
// ---------------------------------------------------------------------------
__global__ void k_softmax_row(const float* __restrict__ W, float* __restrict__ out, int ncol) {
    const int row = blockIdx.x * blockDim.x + threadIdx.x;
    if (row >= WIDTH) return;
    const float* p = W + row * ncol;
    float m = -INFINITY;
    for (int c = 0; c < ncol; ++c) m = fmaxf(m, p[c]);
    float s = 0.f;
    for (int c = 0; c < ncol; ++c) s += __expf(p[c] - m);
    const float inv = 1.0f / s;
    for (int c = 0; c < ncol; ++c) out[row * ncol + c] = __expf(p[c] - m) * inv;
}

// ---------------------------------------------------------------------------
// arg1 = softmax(A1W,0)^T @ inputs ; arg2 likewise (f32, precision-critical)
// ---------------------------------------------------------------------------
__global__ void k_compute_args(const float* __restrict__ a1s, const float* __restrict__ a2s,
                               const float* __restrict__ inp,
                               float* __restrict__ arg1, float* __restrict__ arg2) {
    const int n = blockIdx.x * blockDim.x + threadIdx.x;
    const int w = blockIdx.y;
    float s1 = 0.f, s2 = 0.f;
    for (int j = 0; j < 128; ++j) {
        const float x = inp[j * N_NODE + n];
        s1 += a1s[j * 128 + w] * x;
        s2 += a2s[j * 128 + w] * x;
    }
    arg1[w * N_NODE + n] = s1;
    arg2[w * N_NODE + n] = s2;
}

// ---------------------------------------------------------------------------
// Build bf16 A matrices for the chain GEMM.
// ---------------------------------------------------------------------------
__global__ void k_prep_A(const float* __restrict__ cw, const float* __restrict__ arg2,
                         __bf16* __restrict__ Afwd, __bf16* __restrict__ Abwd) {
    const int idx = blockIdx.x * blockDim.x + threadIdx.x;
    const int w   = blockIdx.y;
    const int r   = idx >> 10;
    const int a   = idx & 1023;
    const float x = arg2[w * N_NODE + a];
    Afwd[(size_t)w * 12288 + idx] = (__bf16)(cw[w * 24 + r] * x);
    Abwd[(size_t)w * 12288 + idx] = (__bf16)(cw[w * 24 + 12 + r] * x);
}

// ---------------------------------------------------------------------------
// Chain GEMM. 96 double-buffered stages of K=128 (4 sub-steps of 32).
// B tiles staged once per block via global_load_async_to_lds_b128; branch-free
// (cndmask) staging addresses; two independent WMMA accumulator chains.
// ---------------------------------------------------------------------------
#define NSTAGE 96   // 12 relations * 8 stages of K=128

__global__ void k_chain_wmma(const __bf16* __restrict__ Afwd, const __bf16* __restrict__ Abwd,
                             const float* __restrict__ DB,
                             const float* __restrict__ arg1, const float* __restrict__ arg2,
                             const float* __restrict__ opw, float* __restrict__ out) {
    // fwd: [stage][sub][a(32)][n(16)+pad] rows 80B (16B-aligned async writes)
    __shared__ __align__(16) float ldsF[2][4][32][20];
    // bwd: [stage][sub][n(16)][a(32)+pad] rows 144B (conflict-free float4 reads)
    __shared__ __align__(16) float ldsB[2][4][16][36];

    const int t      = threadIdx.x;
    const int wave   = t >> 5;
    const int lane   = t & 31;
    const int laneHi = lane >> 4;
    const int mrow   = lane & 15;

    const int wt     = wave;              // M tile (8 per block)
    const int btbase = blockIdx.x * 16;   // N tile base (64 blocks)

    const int wrow = wt * 16 + mrow;
    const __bf16* Af = Afwd + (size_t)wrow * 12288 + 8 * laneHi;
    const __bf16* Ab = Abwd + (size_t)wrow * 12288 + 8 * laneHi;

    // branch-free staging geometry (both variants computed, selected by value)
    const bool isF    = (t < 128);
    const int  a_off  = (t & 127) >> 2;          // fwd: 0..31
    const int  nq     = (t & 3) << 2;            // fwd: 0,4,8,12
    const int  n_b    = (t & 127) >> 3;          // bwd: 0..15
    const int  aq     = (t & 7) << 2;            // bwd: 0,4,...,28

    v8f accF = {}, accB = {};

    auto stage_issue = [&](int stage) {
        const int s     = stage & 1;
        const int r     = stage >> 3;            // 8 stages per relation
        const int ksbeg = (stage & 7) << 7;      // K base within relation
#pragma unroll
        for (int sub = 0; sub < 4; ++sub) {
            const int kb = ksbeg + (sub << 5);
            const size_t offF = (size_t)(kb + a_off) * 1024 + (size_t)(btbase + nq);
            const size_t offB = (size_t)(btbase + n_b) * 1024 + (size_t)(kb + aq);
            const float* g = DB + ((size_t)r << 20) + (isF ? offF : offB);
            const unsigned ldsOff = isF
                ? (unsigned)(unsigned long long)&ldsF[s][sub][a_off][nq]
                : (unsigned)(unsigned long long)&ldsB[s][sub][n_b][aq];
            asm volatile("global_load_async_to_lds_b128 %0, %1, off"
                         :: "v"(ldsOff), "v"(g) : "memory");
        }
    };

    auto compute = [&](int stage) {
        const int s = stage & 1;
#pragma unroll
        for (int sub = 0; sub < 4; ++sub) {
            const int koff = ((stage << 2) + sub) << 5;   // linear K offset

            union { v16bf v; v8bf h[2]; } aF, aB;
            aF.h[0] = *(const v8bf*)(Af + koff);
            aF.h[1] = *(const v8bf*)(Af + koff + 16);
            aB.h[0] = *(const v8bf*)(Ab + koff);
            aB.h[1] = *(const v8bf*)(Ab + koff + 16);

            v16bf bF, bB;
#pragma unroll
            for (int j = 0; j < 16; ++j)      // elem j: K = (j&7) + 8*laneHi + 16*(j>>3), col = mrow
                bF[j] = (__bf16)ldsF[s][sub][(j & 7) + 8 * laneHi + 16 * (j >> 3)][mrow];

            const float4* lp = (const float4*)&ldsB[s][sub][mrow][8 * laneHi];
#pragma unroll
            for (int q = 0; q < 4; ++q) {
                const float4 f = lp[(q & 1) + 4 * (q >> 1)];
                bB[4 * q + 0] = (__bf16)f.x;
                bB[4 * q + 1] = (__bf16)f.y;
                bB[4 * q + 2] = (__bf16)f.z;
                bB[4 * q + 3] = (__bf16)f.w;
            }

            accF = __builtin_amdgcn_wmma_f32_16x16x32_bf16(false, aF.v, false, bF,
                                                           (short)0, accF, false, false);
            accB = __builtin_amdgcn_wmma_f32_16x16x32_bf16(false, aB.v, false, bB,
                                                           (short)0, accB, false, false);
        }
    };

    stage_issue(0);
    stage_issue(1);

    for (int stage = 0; stage < NSTAGE; ++stage) {
        if (stage < NSTAGE - 1) {
            asm volatile("s_wait_asynccnt 0x4" ::: "memory");  // this stage landed,
        } else {                                               // next may be in flight
            asm volatile("s_wait_asynccnt 0x0" ::: "memory");
        }
        __syncthreads();
        compute(stage);
        __syncthreads();
        if (stage + 2 < NSTAGE) stage_issue(stage + 2);
    }

    // Fused epilogue. D element i -> row M = i + 8*laneHi, col N = mrow.
#pragma unroll
    for (int i = 0; i < 8; ++i) {
        const int w   = wt * 16 + i + 8 * laneHi;
        const int col = btbase + mrow;
        const float chain = 1.0f - __expf(-(accF[i] + accB[i]));
        const float a1 = arg1[w * N_NODE + col];
        const float a2 = arg2[w * N_NODE + col];
        const float* o = opw + w * 5;
        const float av = a1 * a2;
        out[w * N_NODE + col] = o[0] * a2 + o[1] * av + o[2] * (a1 + a2 - av)
                              + o[3] * chain + o[4] * (1.0f - a1);
    }
}

// ---------------------------------------------------------------------------
extern "C" void kernel_launch(void* const* d_in, const int* in_sizes, int n_in,
                              void* d_out, int out_size, void* d_ws, size_t ws_size,
                              hipStream_t stream) {
    (void)in_sizes; (void)n_in; (void)out_size; (void)ws_size;

    const float* inputs   = (const float*)d_in[0];   // (128, 1024)
    const float* database = (const float*)d_in[1];   // (12, 1024, 1024)
    const float* a1w      = (const float*)d_in[2];   // (128, 128)
    const float* a2w      = (const float*)d_in[3];   // (128, 128)
    const float* opwRaw   = (const float*)d_in[4];   // (128, 5)
    const float* chainW   = (const float*)d_in[5];   // (128, 24)
    float* out = (float*)d_out;                      // (128, 1024)

    char* p = (char*)d_ws;
    auto carve = [&](size_t bytes) { void* r = (void*)p; p += (bytes + 255) & ~(size_t)255; return r; };
    float*  a1s  = (float*) carve(128 * 128 * sizeof(float));
    float*  a2s  = (float*) carve(128 * 128 * sizeof(float));
    float*  opw  = (float*) carve(128 * 5   * sizeof(float));
    float*  cw   = (float*) carve(128 * 24  * sizeof(float));
    float*  arg1 = (float*) carve((size_t)128 * 1024 * sizeof(float));
    float*  arg2 = (float*) carve((size_t)128 * 1024 * sizeof(float));
    __bf16* Afwd = (__bf16*)carve((size_t)128 * 12288 * sizeof(__bf16));
    __bf16* Abwd = (__bf16*)carve((size_t)128 * 12288 * sizeof(__bf16));

    k_softmax_col128<<<128, 128, 0, stream>>>(a1w, a1s);
    k_softmax_col128<<<128, 128, 0, stream>>>(a2w, a2s);
    k_softmax_row<<<1, 128, 0, stream>>>(opwRaw, opw, 5);
    k_softmax_row<<<1, 128, 0, stream>>>(chainW, cw, 24);

    k_compute_args<<<dim3(4, 128), 256, 0, stream>>>(a1s, a2s, inputs, arg1, arg2);
    k_prep_A<<<dim3(48, 128), 256, 0, stream>>>(cw, arg2, Afwd, Abwd);

    // 64 blocks x 8 waves = 512 16x16 C tiles
    k_chain_wmma<<<64, 256, 0, stream>>>(Afwd, Abwd, database, arg1, arg2, opw, out);
}
